// VanillaAttention_33878702031092
// MI455X (gfx1250) — matrix operand order repaired and checked by reading, not compiled, not run on hardware
//
#include <hip/hip_runtime.h>

#define Bb  2
#define Ss  2048
#define Dd  1024
#define Hh  16
#define HDd 64
#define Mm  (Bb * Ss)
#define ATTN_SCALE 0.125f   // 1/sqrt(64)

typedef __attribute__((ext_vector_type(16))) __bf16 v16bf;
typedef __attribute__((ext_vector_type(8)))  __bf16 v8bf;
typedef __attribute__((ext_vector_type(8)))  float  v8f;

__device__ __forceinline__ v8f wmma_bf16(v16bf a, v16bf b, v8f c) {
  // D = A(16x32 bf16) * B(32x16 bf16) + C(16x16 f32)
  return __builtin_amdgcn_wmma_f32_16x16x32_bf16(false, a, false, b,
                                                 (short)0, c, false, false);
}

__device__ __forceinline__ v16bf cat8(v8bf lo, v8bf hi) {
  return __builtin_shufflevector(lo, hi, 0,1,2,3,4,5,6,7,8,9,10,11,12,13,14,15);
}

// Async copy 16 bytes global -> LDS (tracked by ASYNCcnt).
// VDST = VGPR holding LDS byte address, VADDR = 64-bit global address, GV mode.
__device__ __forceinline__ void async_load16(unsigned lds_addr, const void* g) {
  asm volatile("global_load_async_to_lds_b128 %0, %1, off"
               :: "v"(lds_addr), "v"((unsigned long long)(uintptr_t)g)
               : "memory");
}

// ---------------------------------------------------------------------------
// One-shot f32 -> bf16 conversion (keeps all cvt work out of GEMM K-loops,
// halves GEMM load bytes). 8 elements / thread.
// ---------------------------------------------------------------------------
__global__ __launch_bounds__(256) void cvt_bf16_kernel(
    const float* __restrict__ src, __bf16* __restrict__ dst, int n8) {
  const int i = blockIdx.x * 256 + threadIdx.x;
  if (i < n8) {
    v8f f = *(const v8f*)(src + (size_t)i * 8);
    v8bf o;
#pragma unroll
    for (int j = 0; j < 8; ++j) o[j] = (__bf16)f[j];
    *(v8bf*)(dst + (size_t)i * 8) = o;
  }
}

// ---------------------------------------------------------------------------
// Q/K/V projections on bf16 operands: Y = X @ W^T.
// One wave -> 32x64 tile (2 A-frags x 4 B-frags -> 8 WMMA / K-block; each
// B-fragment reused twice). Q,K stored bf16 [B,H,S,HD]; V stored transposed
// [B,H,HD,S] so the attention P*V B-fragment is a contiguous load.
// ---------------------------------------------------------------------------
__global__ __launch_bounds__(256) void qkv_proj_kernel(
    const __bf16* __restrict__ Xq, const __bf16* __restrict__ Xk,
    const __bf16* __restrict__ Xv,
    const __bf16* __restrict__ Wqb, const __bf16* __restrict__ Wkb,
    const __bf16* __restrict__ Wvb,
    __bf16* __restrict__ Qbf, __bf16* __restrict__ Kbf,
    __bf16* __restrict__ Vt) {
  const int wave = threadIdx.x >> 5;
  const int lane = threadIdx.x & 31;
  const int g = lane >> 4, r = lane & 15;
  const int m0 = blockIdx.x * 256 + wave * 32;
  const int n0 = blockIdx.y * 64;
  const int which = blockIdx.z;

  const __bf16* X = (which == 0) ? Xq  : (which == 1) ? Xk  : Xv;
  const __bf16* W = (which == 0) ? Wqb : (which == 1) ? Wkb : Wvb;

  v8f acc[2][4] = {};
  const __bf16* arow0 = X + (size_t)(m0 + r) * Dd;        // rows m0..m0+15
  const __bf16* arow1 = arow0 + 16 * Dd;                   // rows m0+16..+31

#pragma unroll 1
  for (int k0 = 0; k0 < Dd; k0 += 32) {
    v16bf a0 = cat8(*(const v8bf*)(arow0 + k0 + g * 16),
                    *(const v8bf*)(arow0 + k0 + 8 + g * 16));
    v16bf a1 = cat8(*(const v8bf*)(arow1 + k0 + g * 16),
                    *(const v8bf*)(arow1 + k0 + 8 + g * 16));
#pragma unroll
    for (int t = 0; t < 4; ++t) {
      v16bf b = *(const v16bf*)(W + (size_t)(n0 + t * 16 + r) * Dd
                                  + k0 + 16 * g);
      acc[0][t] = wmma_bf16(a0, b, acc[0][t]);
      acc[1][t] = wmma_bf16(a1, b, acc[1][t]);
    }
  }

  const int bidx = m0 >> 11;          // batch (S = 2048)
  const int h    = n0 >> 6;           // head  (HD = 64)

  if (which == 2) {
#pragma unroll
    for (int u = 0; u < 2; ++u)
#pragma unroll
      for (int t = 0; t < 4; ++t) {
        v8bf vals;
#pragma unroll
        for (int j = 0; j < 8; ++j) vals[j] = (__bf16)acc[u][t][j];
        const int s = (m0 & (Ss - 1)) + u * 16 + g * 8;
        __bf16* dst = Vt + (((size_t)bidx * Hh + h) * HDd + (t * 16 + r)) * Ss
                         + s;
        *(v8bf*)dst = vals;
      }
  } else {
    __bf16* Out = (which == 0) ? Qbf : Kbf;
#pragma unroll
    for (int u = 0; u < 2; ++u)
#pragma unroll
      for (int t = 0; t < 4; ++t)
#pragma unroll
        for (int j = 0; j < 8; ++j) {
          const int s = (m0 & (Ss - 1)) + u * 16 + j + 8 * g;
          Out[(((size_t)bidx * Hh + h) * Ss + s) * HDd + t * 16 + r] =
              (__bf16)acc[u][t][j];
        }
  }
}

// ---------------------------------------------------------------------------
// Flash attention. All 8 waves of a block share one (b,h) and walk the same
// K/V stream, so K (32x64) and V^T (64x32) blocks are staged into LDS once
// per workgroup with global_load_async_to_lds_b128, double-buffered on
// ASYNCcnt. Wave fragment reads become ds_load_b128. Online softmax with the
// P tile bounced through a per-wave LDS region (C-layout -> A-layout).
// ---------------------------------------------------------------------------
__global__ __launch_bounds__(256) void attn_kernel(
    const __bf16* __restrict__ Qbf, const __bf16* __restrict__ Kbf,
    const __bf16* __restrict__ Vt, __bf16* __restrict__ ctx) {
  __shared__ __align__(64) __bf16 Ptile[8][16 * 32];   // 1KB per wave
  __shared__ __align__(64) __bf16 Ktile[2][32 * 64];   // 4KB per buffer
  __shared__ __align__(64) __bf16 Vtile[2][64 * 32];   // 4KB per buffer

  const int wave = threadIdx.x >> 5;
  const int lane = threadIdx.x & 31;
  const int g = lane >> 4, r = lane & 15;
  const int tile = blockIdx.x * 8 + wave;
  const int bh = tile >> 7;           // 128 q-tiles per (b,h); 8 per block
  const int qt = tile & 127;
  const int b = bh >> 4, h = bh & 15;

  const __bf16* qrow  = Qbf + ((size_t)bh * Ss + qt * 16 + r) * HDd;
  const __bf16* kbase = Kbf + (size_t)bh * Ss * HDd;
  const __bf16* vbase = Vt  + (size_t)bh * HDd * Ss;

  // cooperative stage of one 32-key block: 16B of K + 16B of V per thread
  const int ti = threadIdx.x;
  auto stage = [&](int buf, int k0n) {
    unsigned kd = (unsigned)(uintptr_t)(&Ktile[buf][0]) + (unsigned)(ti * 16);
    async_load16(kd, kbase + (size_t)k0n * HDd + ti * 8);
    unsigned vd = (unsigned)(uintptr_t)(&Vtile[buf][0])
                + (unsigned)((ti >> 2) * 64 + (ti & 3) * 16);
    async_load16(vd, vbase + (size_t)(ti >> 2) * Ss + k0n + (ti & 3) * 8);
  };

  // Q A-fragments (HD=64 -> two K=32 fragments), kept in registers
  v16bf qa0 = cat8(*(const v8bf*)(qrow + g * 16),
                   *(const v8bf*)(qrow + 8 + g * 16));
  v16bf qa1 = cat8(*(const v8bf*)(qrow + 32 + g * 16),
                   *(const v8bf*)(qrow + 40 + g * 16));

  float mrun[8], lsum[8];
  v8f oacc[4] = {};
#pragma unroll
  for (int j = 0; j < 8; ++j) { mrun[j] = -1e30f; lsum[j] = 0.f; }

  __bf16* P = &Ptile[wave][0];

  stage(0, 0);   // prologue: first K/V block in flight

#pragma unroll 1
  for (int k0 = 0; k0 < Ss; k0 += 32) {
    const int buf = (k0 >> 5) & 1;
    if (k0 + 32 < Ss) {
      stage(buf ^ 1, k0 + 32);                       // prefetch next block
      asm volatile("s_wait_asynccnt 0x2" ::: "memory");  // current block done
    } else {
      asm volatile("s_wait_asynccnt 0x0" ::: "memory");
    }
    __syncthreads();                                 // all stagers landed

    const __bf16* Kt = &Ktile[buf][0];               // [32 keys][64 hd]
    const __bf16* Vl = &Vtile[buf][0];               // [64 hd][32 keys]

    // ---- scores: S = Q * K^T, two 16x16 tiles from LDS
    v8f s0 = {}, s1 = {};
    {
      const __bf16* krow  = Kt + r * HDd;            // key r (lane = key col)
      const __bf16* krow2 = Kt + (16 + r) * HDd;
      s0 = wmma_bf16(qa0, *(const v16bf*)(krow  + 16 * g),      s0);
      s0 = wmma_bf16(qa1, *(const v16bf*)(krow  + 32 + 16 * g), s0);
      s1 = wmma_bf16(qa0, *(const v16bf*)(krow2 + 16 * g),      s1);
      s1 = wmma_bf16(qa1, *(const v16bf*)(krow2 + 32 + 16 * g), s1);
    }
    // ---- online softmax over the 32 new columns (row = j + 8g, 16 lanes)
#pragma unroll
    for (int j = 0; j < 8; ++j) {
      float x0 = s0[j] * ATTN_SCALE, x1 = s1[j] * ATTN_SCALE;
      float mx = fmaxf(x0, x1);
      mx = fmaxf(mx, __shfl_xor(mx, 8, 16));
      mx = fmaxf(mx, __shfl_xor(mx, 4, 16));
      mx = fmaxf(mx, __shfl_xor(mx, 2, 16));
      mx = fmaxf(mx, __shfl_xor(mx, 1, 16));
      float mnew = fmaxf(mrun[j], mx);
      float corr = __expf(mrun[j] - mnew);
      float e0 = __expf(x0 - mnew);
      float e1 = __expf(x1 - mnew);
      lsum[j] = lsum[j] * corr + e0 + e1;
      mrun[j] = mnew;
#pragma unroll
      for (int t = 0; t < 4; ++t) oacc[t][j] *= corr;
      P[(j + 8 * g) * 32 + r]      = (__bf16)e0;     // C-layout -> row major
      P[(j + 8 * g) * 32 + 16 + r] = (__bf16)e1;
    }
    asm volatile("s_wait_dscnt 0" ::: "memory");
    // ---- reload P as A-fragment (row = lane, K = 32 keys)
    v16bf pa = cat8(*(const v8bf*)(P + r * 32 + g * 16),
                    *(const v8bf*)(P + r * 32 + 8 + g * 16));
    // ---- O += P * V from the staged V^T tile (contiguous 32B / lane)
#pragma unroll
    for (int t = 0; t < 4; ++t) {
      const __bf16* vrow = Vl + (t * 16 + r) * 32 + 16 * g;
      oacc[t] = wmma_bf16(pa, *(const v16bf*)vrow, oacc[t]);
    }
    __syncthreads();   // everyone done reading buf before it is re-staged
  }

  // ---- normalize (reduce distributed row sums across the 16 lanes)
#pragma unroll
  for (int j = 0; j < 8; ++j) {
    float s = lsum[j];
    s += __shfl_xor(s, 8, 16);
    s += __shfl_xor(s, 4, 16);
    s += __shfl_xor(s, 2, 16);
    s += __shfl_xor(s, 1, 16);
    float inv = 1.0f / s;
#pragma unroll
    for (int t = 0; t < 4; ++t) oacc[t][j] *= inv;
  }

  // ---- merged-head store: ctx bf16 [B,S,D]
  __bf16* cbase = ctx + ((size_t)b * Ss + qt * 16) * Dd + h * HDd;
#pragma unroll
  for (int t = 0; t < 4; ++t)
#pragma unroll
    for (int j = 0; j < 8; ++j)
      cbase[(size_t)(j + 8 * g) * Dd + t * 16 + r] = (__bf16)oacc[t][j];
}

// ---------------------------------------------------------------------------
// Output projection: out = ctx(bf16) @ Wo^T(bf16) -> f32, 32x64 wave tile.
// ---------------------------------------------------------------------------
__global__ __launch_bounds__(256) void out_proj_kernel(
    const __bf16* __restrict__ ctx, const __bf16* __restrict__ Wob,
    float* __restrict__ out) {
  const int wave = threadIdx.x >> 5;
  const int lane = threadIdx.x & 31;
  const int g = lane >> 4, r = lane & 15;
  const int m0 = blockIdx.x * 256 + wave * 32;
  const int n0 = blockIdx.y * 64;

  v8f acc[2][4] = {};
  const __bf16* arow0 = ctx + (size_t)(m0 + r) * Dd;
  const __bf16* arow1 = arow0 + 16 * Dd;

#pragma unroll 1
  for (int k0 = 0; k0 < Dd; k0 += 32) {
    v16bf a0 = cat8(*(const v8bf*)(arow0 + k0 + g * 16),
                    *(const v8bf*)(arow0 + k0 + 8 + g * 16));
    v16bf a1 = cat8(*(const v8bf*)(arow1 + k0 + g * 16),
                    *(const v8bf*)(arow1 + k0 + 8 + g * 16));
#pragma unroll
    for (int t = 0; t < 4; ++t) {
      v16bf bfr = *(const v16bf*)(Wob + (size_t)(n0 + t * 16 + r) * Dd
                                      + k0 + 16 * g);
      acc[0][t] = wmma_bf16(a0, bfr, acc[0][t]);
      acc[1][t] = wmma_bf16(a1, bfr, acc[1][t]);
    }
  }
#pragma unroll
  for (int u = 0; u < 2; ++u)
#pragma unroll
    for (int t = 0; t < 4; ++t)
#pragma unroll
      for (int j = 0; j < 8; ++j)
        out[(size_t)(m0 + u * 16 + j + 8 * g) * Dd + n0 + t * 16 + r] =
            acc[u][t][j];
}

// ---------------------------------------------------------------------------
extern "C" void kernel_launch(void* const* d_in, const int* in_sizes, int n_in,
                              void* d_out, int out_size, void* d_ws,
                              size_t ws_size, hipStream_t stream) {
  (void)in_sizes; (void)n_in; (void)out_size; (void)ws_size;
  const float* q  = (const float*)d_in[0];
  const float* k  = (const float*)d_in[1];
  const float* v  = (const float*)d_in[2];
  const float* Wq = (const float*)d_in[3];
  const float* Wk = (const float*)d_in[4];
  const float* Wv = (const float*)d_in[5];
  const float* Wo = (const float*)d_in[6];

  const size_t MD = (size_t)Mm * Dd;   // 4M elems
  const size_t DD = (size_t)Dd * Dd;   // 1M elems

  __bf16* Xq  = (__bf16*)d_ws;         // bf16 copies of inputs
  __bf16* Xk  = Xq  + MD;
  __bf16* Xv  = Xk  + MD;
  __bf16* Wqb = Xv  + MD;              // bf16 copies of weights
  __bf16* Wkb = Wqb + DD;
  __bf16* Wvb = Wkb + DD;
  __bf16* Wob = Wvb + DD;
  __bf16* Qbf = Wob + DD;              // projected Q [B,H,S,HD]
  __bf16* Kbf = Qbf + MD;              // projected K [B,H,S,HD]
  __bf16* Vt  = Kbf + MD;              // projected V transposed [B,H,HD,S]
  __bf16* ctx = Vt  + MD;              // attention output [B,S,D]

  dim3 blk(256, 1, 1);

  cvt_bf16_kernel<<<dim3(MD / 2048), blk, 0, stream>>>(q,  Xq,  (int)(MD / 8));
  cvt_bf16_kernel<<<dim3(MD / 2048), blk, 0, stream>>>(k,  Xk,  (int)(MD / 8));
  cvt_bf16_kernel<<<dim3(MD / 2048), blk, 0, stream>>>(v,  Xv,  (int)(MD / 8));
  cvt_bf16_kernel<<<dim3(DD / 2048), blk, 0, stream>>>(Wq, Wqb, (int)(DD / 8));
  cvt_bf16_kernel<<<dim3(DD / 2048), blk, 0, stream>>>(Wk, Wkb, (int)(DD / 8));
  cvt_bf16_kernel<<<dim3(DD / 2048), blk, 0, stream>>>(Wv, Wvb, (int)(DD / 8));
  cvt_bf16_kernel<<<dim3(DD / 2048), blk, 0, stream>>>(Wo, Wob, (int)(DD / 8));

  qkv_proj_kernel<<<dim3(Mm / 256, Dd / 64, 3), blk, 0, stream>>>(
      Xq, Xk, Xv, Wqb, Wkb, Wvb, Qbf, Kbf, Vt);
  attn_kernel<<<dim3((Bb * Hh * (Ss / 16)) / 8, 1, 1), blk, 0, stream>>>(
      Qbf, Kbf, Vt, ctx);
  out_proj_kernel<<<dim3(Mm / 256, Dd / 64, 1), blk, 0, stream>>>(
      ctx, Wob, (float*)d_out);
}
